// TransformerEncoderLayer_84198538871067
// MI455X (gfx1250) — matrix operand order repaired and checked by reading, not compiled, not run on hardware
//
#include <hip/hip_runtime.h>
#include <hip/hip_bf16.h>

// ---------------------------------------------------------------------------
// Transformer encoder layer on gfx1250 (MI455X): bf16 WMMA everywhere,
// async global->LDS staging, DPP softmax reductions.
// S=2048 B=4 D=1024 H=16 DH=64 FF=4096
// ---------------------------------------------------------------------------

#define S_LEN  2048
#define BATCH  4
#define DMODEL 1024
#define NHEAD  16
#define DHEAD  64
#define FFDIM  4096
#define ROWS   (S_LEN * BATCH)   // 8192 token rows

typedef __bf16 bf16_t;
typedef __attribute__((ext_vector_type(16))) __bf16 bf16x16;
typedef __attribute__((ext_vector_type(8)))  float  f32x8;
typedef __attribute__((ext_vector_type(4)))  int    i32x4;

__device__ __forceinline__ f32x8 f32x8_zero() {
    f32x8 z;
#pragma unroll
    for (int i = 0; i < 8; ++i) z[i] = 0.0f;
    return z;
}

// ---------------------------------------------------------------------------
// gfx1250 async global->LDS copy (16B per lane), ASYNCcnt-tracked.
// clang signature: (v4i32 AS1*, v4i32 AS3*, imm offset, imm cpol).
// Falls back to a synchronous VGPR round trip if the builtin is absent.
// ---------------------------------------------------------------------------
#if defined(__AMDGCN__) && __has_builtin(__builtin_amdgcn_global_load_async_to_lds_b128)
#define HAVE_ASYNC_LDS 1
#else
#define HAVE_ASYNC_LDS 0
#endif

typedef __attribute__((address_space(1))) i32x4 as1_i32x4;
typedef __attribute__((address_space(3))) i32x4 as3_i32x4;

__device__ __forceinline__ void async_copy16(const void* g, void* l) {
#if HAVE_ASYNC_LDS
    __builtin_amdgcn_global_load_async_to_lds_b128(
        (as1_i32x4*)g, (as3_i32x4*)l, 0, 0);
#else
    *(uint4*)l = *(const uint4*)g;
#endif
}

__device__ __forceinline__ void wait_async_lds() {
#if HAVE_ASYNC_LDS
#if __has_builtin(__builtin_amdgcn_s_wait_asynccnt)
    __builtin_amdgcn_s_wait_asynccnt(0);
#else
    asm volatile("s_wait_asynccnt 0x0" ::: "memory");
#endif
#endif
}

// ---------------------------------------------------------------------------
// 16-lane (DPP row) butterfly reductions: rows of a WMMA C tile live across
// 16 lanes of one VGPR, so quad_perm xor1/xor2 + half_mirror + mirror reduce
// the row entirely in VALU (no LDS bpermute traffic).
// ---------------------------------------------------------------------------
#if defined(__AMDGCN__) && __has_builtin(__builtin_amdgcn_update_dpp)
#define ROW16_DPP(v, expr_combine)                                              \
    do {                                                                        \
        float _o;                                                               \
        _o = __int_as_float(__builtin_amdgcn_update_dpp(                        \
            0, __float_as_int(v), 0xB1, 0xF, 0xF, true));  v = expr_combine;    \
        _o = __int_as_float(__builtin_amdgcn_update_dpp(                        \
            0, __float_as_int(v), 0x4E, 0xF, 0xF, true));  v = expr_combine;    \
        _o = __int_as_float(__builtin_amdgcn_update_dpp(                        \
            0, __float_as_int(v), 0x141, 0xF, 0xF, true)); v = expr_combine;    \
        _o = __int_as_float(__builtin_amdgcn_update_dpp(                        \
            0, __float_as_int(v), 0x140, 0xF, 0xF, true)); v = expr_combine;    \
    } while (0)

__device__ __forceinline__ float row16_max(float v) {
    ROW16_DPP(v, fmaxf(v, _o));
    return v;
}
__device__ __forceinline__ float row16_sum(float v) {
    ROW16_DPP(v, v + _o);
    return v;
}
#else
__device__ __forceinline__ float row16_max(float v) {
    v = fmaxf(v, __shfl_xor(v, 1));
    v = fmaxf(v, __shfl_xor(v, 2));
    v = fmaxf(v, __shfl_xor(v, 4));
    v = fmaxf(v, __shfl_xor(v, 8));
    return v;
}
__device__ __forceinline__ float row16_sum(float v) {
    v += __shfl_xor(v, 1);
    v += __shfl_xor(v, 2);
    v += __shfl_xor(v, 4);
    v += __shfl_xor(v, 8);
    return v;
}
#endif

// ---------------------------------------------------------------------------
// f32 -> bf16 conversion (vectorized)
// ---------------------------------------------------------------------------
__global__ __launch_bounds__(256)
void cvt_f32_bf16(const float* __restrict__ in, bf16_t* __restrict__ out, int n) {
    int i = (blockIdx.x * 256 + threadIdx.x) * 4;
    if (i < n) {  // n is always a multiple of 4 here
        float4 v = *(const float4*)(in + i);
        out[i + 0] = (bf16_t)v.x;
        out[i + 1] = (bf16_t)v.y;
        out[i + 2] = (bf16_t)v.z;
        out[i + 3] = (bf16_t)v.w;
    }
}

// ---------------------------------------------------------------------------
// bf16 WMMA GEMM: C[M,N] = A[M,K] @ B[K,N] + bias, optional ReLU.
// Block tile 128x128, 8 waves (wave32), each wave 32x64 (2x4 WMMA tiles).
// A tile staged with async b128 copies (ASYNCcnt); B staged transposed via
// VGPRs so both A and B fragment loads are contiguous bf16 pairs.
// ---------------------------------------------------------------------------
template<bool RELU, bool OUT_F32, bool OUT_BF16>
__global__ __launch_bounds__(256)
void gemm_bf16(const bf16_t* __restrict__ A, const bf16_t* __restrict__ Bw,
               const float* __restrict__ bias,
               float* __restrict__ Cf, bf16_t* __restrict__ Cb,
               int M, int N, int K) {
    constexpr int LDA = 40;   // 32 + 8 pad (elements)
    constexpr int LDB = 40;   // transposed B: [n][k], 32 + 8 pad
    __shared__ __align__(16) bf16_t sA[128 * LDA];
    __shared__ __align__(16) bf16_t sB[128 * LDB];

    const int tid  = threadIdx.x;
    const int wid  = tid >> 5;
    const int lane = tid & 31;
    const int wm   = wid & 3;      // wave row (4 waves along M -> 128 rows)
    const int wn   = wid >> 2;     // wave col (2 waves along N -> 128 cols)
    const int row0 = blockIdx.y * 128;
    const int col0 = blockIdx.x * 128;
    const int lr   = lane & 15;
    const int kh   = lane >> 4;

    f32x8 acc[2][4];
#pragma unroll
    for (int mi = 0; mi < 2; ++mi)
#pragma unroll
        for (int ni = 0; ni < 4; ++ni) acc[mi][ni] = f32x8_zero();

    // staging coordinates
    const int ar = tid >> 2;            // 0..63 (A row)
    const int ac = (tid & 3) * 8;       // 0,8,16,24
    const int bkr = tid >> 4;           // 0..15 (B k row)
    const int bcb = (tid & 15) * 8;     // 0..120

    for (int k0 = 0; k0 < K; k0 += 32) {
        __syncthreads();
        // ---- stage A tile: 128 rows x 32 k, async global->LDS (b128/lane)
#pragma unroll
        for (int it = 0; it < 2; ++it) {
            const bf16_t* gp = A + (size_t)(row0 + ar + it * 64) * K + k0 + ac;
            async_copy16(gp, &sA[(ar + it * 64) * LDA + ac]);
        }
        // ---- stage B tile transposed: global [k][n] -> LDS [n][k]
#pragma unroll
        for (int it = 0; it < 2; ++it) {
            const bf16_t* gp = Bw + (size_t)(k0 + bkr + it * 16) * N + col0 + bcb;
            if (k0 + 32 < K)  // pull next slab toward L2/WGP$
                __builtin_prefetch(gp + (size_t)32 * N, 0, 1);
            uint4 v = *(const uint4*)gp;
            bf16_t tmp[8];
            *(uint4*)tmp = v;
#pragma unroll
            for (int j = 0; j < 8; ++j)
                sB[(bcb + j) * LDB + (bkr + it * 16)] = tmp[j];
        }
        wait_async_lds();
        __syncthreads();

        // ---- fragment loads (contiguous k = 16*kh + s in both A and B)
        bf16x16 af[2];
#pragma unroll
        for (int mi = 0; mi < 2; ++mi) {
            int rb = (wm * 32 + mi * 16 + lr) * LDA;
            bf16x16 a;
#pragma unroll
            for (int i = 0; i < 8; ++i) {
                int k = kh * 16 + 2 * i;
                a[2 * i]     = sA[rb + k];
                a[2 * i + 1] = sA[rb + k + 1];
            }
            af[mi] = a;
        }
        bf16x16 bfr[4];
#pragma unroll
        for (int ni = 0; ni < 4; ++ni) {
            int cb2 = (wn * 64 + ni * 16 + lr) * LDB + kh * 16;
            bf16x16 b;
#pragma unroll
            for (int i = 0; i < 8; ++i) {
                b[2 * i]     = sB[cb2 + 2 * i];
                b[2 * i + 1] = sB[cb2 + 2 * i + 1];
            }
            bfr[ni] = b;
        }
#pragma unroll
        for (int mi = 0; mi < 2; ++mi)
#pragma unroll
            for (int ni = 0; ni < 4; ++ni)
                acc[mi][ni] = __builtin_amdgcn_wmma_f32_16x16x32_bf16(
                    false, af[mi], false, bfr[ni], (short)0, acc[mi][ni], false, false);
    }

    // ---- epilogue: bias (+ReLU), write f32 and/or bf16
#pragma unroll
    for (int mi = 0; mi < 2; ++mi) {
#pragma unroll
        for (int ni = 0; ni < 4; ++ni) {
            int col = col0 + wn * 64 + ni * 16 + lr;
            float bs = bias ? bias[col] : 0.0f;
#pragma unroll
            for (int i = 0; i < 8; ++i) {
                int row = row0 + wm * 32 + mi * 16 + i + kh * 8;
                float v = acc[mi][ni][i] + bs;
                if (RELU) v = fmaxf(v, 0.0f);
                size_t idx = (size_t)row * N + col;
                if (OUT_F32)  Cf[idx] = v;
                if (OUT_BF16) Cb[idx] = (bf16_t)v;
            }
        }
    }
}

// ---------------------------------------------------------------------------
// Flash attention: grid = (S/128, B*H). 8 waves; each wave owns 16 q rows x
// DH=64 in registers. Key blocks of 32 staged to LDS (K async, V transposed).
// QK^T and P@V via bf16 WMMA; online softmax in f32 with DPP row reductions.
// ---------------------------------------------------------------------------
__global__ __launch_bounds__(256)
void flash_attn(const bf16_t* __restrict__ Q, const bf16_t* __restrict__ K,
                const bf16_t* __restrict__ V, bf16_t* __restrict__ ctx) {
    constexpr int LKT = 72;   // sK: [t(32)][dh(64)] padded
    constexpr int LVT = 40;   // sV: [dh(64)][t(32)] padded (transposed)
    constexpr int LPP = 40;   // per-wave P tile: [r(16)][t(32)] padded
    __shared__ __align__(16) bf16_t sK[32 * LKT];
    __shared__ __align__(16) bf16_t sV[64 * LVT];
    __shared__ __align__(16) bf16_t sP[8][16 * LPP];

    const int tid = threadIdx.x;
    const int wid = tid >> 5, lane = tid & 31;
    const int lr  = lane & 15, kh = lane >> 4;
    const int bh  = blockIdx.y;
    const int b   = bh >> 4;        // H = 16
    const int h   = bh & 15;
    const int q0  = blockIdx.x * 128 + wid * 16;
    const float scale = 0.125f;     // 1/sqrt(64)

    // ---- load Q fragments (A layout), two 32-deep k chunks over DH=64
    bf16x16 aQ[2];
#pragma unroll
    for (int c = 0; c < 2; ++c) {
        const bf16_t* qp = Q + ((size_t)(q0 + lr) * BATCH + b) * DMODEL + h * DHEAD + c * 32;
        bf16x16 a;
#pragma unroll
        for (int i = 0; i < 8; ++i) {
            int k = kh * 16 + 2 * i;
            a[2 * i]     = qp[k];
            a[2 * i + 1] = qp[k + 1];
        }
        aQ[c] = a;
    }

    f32x8 o[4];
#pragma unroll
    for (int ni = 0; ni < 4; ++ni) o[ni] = f32x8_zero();
    float m[8], l[8];
#pragma unroll
    for (int i = 0; i < 8; ++i) { m[i] = -1e30f; l[i] = 0.0f; }

    const int st_t = tid >> 3;          // 0..31 (staging key row)
    const int st_c = (tid & 7) * 8;     // 0..56 (staging dh base)

    for (int kt = 0; kt < S_LEN; kt += 32) {
        __syncthreads();
        // ---- stage K block row-major (async b128) and V block transposed
        {
            const bf16_t* kp = K + ((size_t)(kt + st_t) * BATCH + b) * DMODEL + h * DHEAD + st_c;
            async_copy16(kp, &sK[st_t * LKT + st_c]);
            const bf16_t* vp = V + ((size_t)(kt + st_t) * BATCH + b) * DMODEL + h * DHEAD + st_c;
            uint4 vv = *(const uint4*)vp;
            bf16_t tmp[8];
            *(uint4*)tmp = vv;
#pragma unroll
            for (int j = 0; j < 8; ++j) sV[(st_c + j) * LVT + st_t] = tmp[j];
        }
        wait_async_lds();
        __syncthreads();

        // ---- scores: S_blk(16x32) = Q(16x64) @ K_blk^T  -> two 16x16 tiles
        f32x8 st[2];
#pragma unroll
        for (int ni = 0; ni < 2; ++ni) {
            int t = ni * 16 + lr;
            bf16x16 b0, b1;
#pragma unroll
            for (int i = 0; i < 8; ++i) {
                int k = kh * 16 + 2 * i;
                b0[2 * i]     = sK[t * LKT + k];
                b0[2 * i + 1] = sK[t * LKT + k + 1];
                b1[2 * i]     = sK[t * LKT + 32 + k];
                b1[2 * i + 1] = sK[t * LKT + 32 + k + 1];
            }
            f32x8 z = f32x8_zero();
            z = __builtin_amdgcn_wmma_f32_16x16x32_bf16(false, aQ[0], false, b0, (short)0, z, false, false);
            z = __builtin_amdgcn_wmma_f32_16x16x32_bf16(false, aQ[1], false, b1, (short)0, z, false, false);
            st[ni] = z;
        }

        // ---- online softmax (each row lives across 16 lanes of one VGPR)
        float alpha[8];
#pragma unroll
        for (int i = 0; i < 8; ++i) {
            float v0 = st[0][i] * scale, v1 = st[1][i] * scale;
            float mx = row16_max(fmaxf(v0, v1));
            float mnew = fmaxf(m[i], mx);
            float a  = __expf(m[i] - mnew);
            float p0 = __expf(v0 - mnew);
            float p1 = __expf(v1 - mnew);
            st[0][i] = p0; st[1][i] = p1;
            float rs = row16_sum(p0 + p1);
            l[i] = l[i] * a + rs;
            m[i] = mnew;
            alpha[i] = a;
        }
#pragma unroll
        for (int ni = 0; ni < 4; ++ni)
#pragma unroll
            for (int i = 0; i < 8; ++i) o[ni][i] *= alpha[i];

        // ---- P: C-layout -> A-layout via per-wave LDS tile (bf16)
#pragma unroll
        for (int ni = 0; ni < 2; ++ni)
#pragma unroll
            for (int i = 0; i < 8; ++i)
                sP[wid][(i + kh * 8) * LPP + ni * 16 + lr] = (bf16_t)st[ni][i];

        bf16x16 aP;
        {
            const bf16_t* pp = &sP[wid][lr * LPP];
#pragma unroll
            for (int i = 0; i < 8; ++i) {
                int k = kh * 16 + 2 * i;
                aP[2 * i]     = pp[k];
                aP[2 * i + 1] = pp[k + 1];
            }
        }

        // ---- O(16x64) += P(16x32) @ V_blk(32x64)
#pragma unroll
        for (int ni = 0; ni < 4; ++ni) {
            int dh = ni * 16 + lr;
            const bf16_t* vp2 = &sV[dh * LVT + kh * 16];
            bf16x16 bv;
#pragma unroll
            for (int i = 0; i < 8; ++i) {
                bv[2 * i]     = vp2[2 * i];
                bv[2 * i + 1] = vp2[2 * i + 1];
            }
            o[ni] = __builtin_amdgcn_wmma_f32_16x16x32_bf16(false, aP, false, bv, (short)0, o[ni], false, false);
        }
    }

    // ---- normalize and write ctx (bf16, feeds the Wo GEMM)
#pragma unroll
    for (int i = 0; i < 8; ++i) l[i] = 1.0f / l[i];
#pragma unroll
    for (int ni = 0; ni < 4; ++ni) {
#pragma unroll
        for (int i = 0; i < 8; ++i) {
            int s = q0 + i + kh * 8;
            ctx[((size_t)s * BATCH + b) * DMODEL + h * DHEAD + ni * 16 + lr] =
                (bf16_t)(o[ni][i] * l[i]);
        }
    }
}

// ---------------------------------------------------------------------------
// Fused residual add + LayerNorm over D=1024. One block per row, 256 threads,
// 4 elements per thread. Optionally writes a bf16 copy for the next GEMM.
// ---------------------------------------------------------------------------
__global__ __launch_bounds__(256)
void add_layernorm(const float* __restrict__ X, const float* __restrict__ Y,
                   const float* __restrict__ g, const float* __restrict__ be,
                   float* __restrict__ out, bf16_t* __restrict__ outb) {
    __shared__ float red[8];
    const int row = blockIdx.x;
    const float* xp = X + (size_t)row * DMODEL;
    const float* yp = Y + (size_t)row * DMODEL;
    const int wid = threadIdx.x >> 5, lane = threadIdx.x & 31;

    float v[4];
    float s = 0.0f;
#pragma unroll
    for (int j = 0; j < 4; ++j) {
        int c = threadIdx.x + j * 256;
        v[j] = xp[c] + yp[c];
        s += v[j];
    }
    s = row16_sum(s);
    s += __shfl_xor(s, 16);
    if (lane == 0) red[wid] = s;
    __syncthreads();
    float tot = 0.0f;
#pragma unroll
    for (int i = 0; i < 8; ++i) tot += red[i];
    const float mu = tot * (1.0f / DMODEL);
    __syncthreads();

    float s2 = 0.0f;
#pragma unroll
    for (int j = 0; j < 4; ++j) {
        float d = v[j] - mu;
        s2 += d * d;
    }
    s2 = row16_sum(s2);
    s2 += __shfl_xor(s2, 16);
    if (lane == 0) red[wid] = s2;
    __syncthreads();
    float tv = 0.0f;
#pragma unroll
    for (int i = 0; i < 8; ++i) tv += red[i];
    const float rstd = rsqrtf(tv * (1.0f / DMODEL) + 1e-5f);

#pragma unroll
    for (int j = 0; j < 4; ++j) {
        int c = threadIdx.x + j * 256;
        float ov = (v[j] - mu) * rstd * g[c] + be[c];
        out[(size_t)row * DMODEL + c] = ov;
        if (outb) outb[(size_t)row * DMODEL + c] = (bf16_t)ov;
    }
}

// ---------------------------------------------------------------------------
// Host orchestration
// ---------------------------------------------------------------------------
extern "C" void kernel_launch(void* const* d_in, const int* in_sizes, int n_in,
                              void* d_out, int out_size, void* d_ws, size_t ws_size,
                              hipStream_t stream) {
    const float* src = (const float*)d_in[0];
    const float* Wq  = (const float*)d_in[1];
    const float* bq  = (const float*)d_in[2];
    const float* Wk  = (const float*)d_in[3];
    const float* bk  = (const float*)d_in[4];
    const float* Wv  = (const float*)d_in[5];
    const float* bv  = (const float*)d_in[6];
    const float* Wo  = (const float*)d_in[7];
    const float* bo  = (const float*)d_in[8];
    const float* W1  = (const float*)d_in[9];
    const float* b1  = (const float*)d_in[10];
    const float* W2  = (const float*)d_in[11];
    const float* b2  = (const float*)d_in[12];
    const float* g1  = (const float*)d_in[13];
    const float* be1 = (const float*)d_in[14];
    const float* g2  = (const float*)d_in[15];
    const float* be2 = (const float*)d_in[16];
    (void)in_sizes; (void)n_in; (void)out_size; (void)ws_size;

    char* ws = (char*)d_ws;
    size_t off = 0;
    auto alloc = [&](size_t bytes) -> void* {
        off = (off + 255) & ~(size_t)255;
        void* p = ws + off;
        off += bytes;
        return p;
    };

    const size_t TOK  = (size_t)ROWS * DMODEL;       // 8192*1024
    const size_t TOKF = (size_t)ROWS * FFDIM;        // 8192*4096
    bf16_t* srcb  = (bf16_t*)alloc(TOK * 2);
    bf16_t* Wqb   = (bf16_t*)alloc((size_t)DMODEL * DMODEL * 2);
    bf16_t* Wkb   = (bf16_t*)alloc((size_t)DMODEL * DMODEL * 2);
    bf16_t* Wvb   = (bf16_t*)alloc((size_t)DMODEL * DMODEL * 2);
    bf16_t* Wob   = (bf16_t*)alloc((size_t)DMODEL * DMODEL * 2);
    bf16_t* W1b   = (bf16_t*)alloc((size_t)DMODEL * FFDIM * 2);
    bf16_t* W2b   = (bf16_t*)alloc((size_t)FFDIM * DMODEL * 2);
    bf16_t* Qb    = (bf16_t*)alloc(TOK * 2);
    bf16_t* Kb    = (bf16_t*)alloc(TOK * 2);
    bf16_t* Vb    = (bf16_t*)alloc(TOK * 2);
    bf16_t* ctxb  = (bf16_t*)alloc(TOK * 2);
    float*  attn  = (float*)alloc(TOK * 4);
    float*  src2  = (float*)alloc(TOK * 4);
    bf16_t* src2b = (bf16_t*)alloc(TOK * 2);
    bf16_t* h1b   = (bf16_t*)alloc(TOKF * 2);
    float*  ff    = (float*)alloc(TOK * 4);

    auto cvt = [&](const float* in, bf16_t* out, size_t n) {
        int blocks = (int)((n / 4 + 255) / 256);
        cvt_f32_bf16<<<blocks, 256, 0, stream>>>(in, out, (int)n);
    };
    cvt(src, srcb, TOK);
    cvt(Wq, Wqb, (size_t)DMODEL * DMODEL);
    cvt(Wk, Wkb, (size_t)DMODEL * DMODEL);
    cvt(Wv, Wvb, (size_t)DMODEL * DMODEL);
    cvt(Wo, Wob, (size_t)DMODEL * DMODEL);
    cvt(W1, W1b, (size_t)DMODEL * FFDIM);
    cvt(W2, W2b, (size_t)FFDIM * DMODEL);

    // Q/K/V projections: (8192x1024)@(1024x1024)+bias -> bf16
    {
        dim3 grid(DMODEL / 128, ROWS / 128);
        gemm_bf16<false, false, true><<<grid, 256, 0, stream>>>(
            srcb, Wqb, bq, nullptr, Qb, ROWS, DMODEL, DMODEL);
        gemm_bf16<false, false, true><<<grid, 256, 0, stream>>>(
            srcb, Wkb, bk, nullptr, Kb, ROWS, DMODEL, DMODEL);
        gemm_bf16<false, false, true><<<grid, 256, 0, stream>>>(
            srcb, Wvb, bv, nullptr, Vb, ROWS, DMODEL, DMODEL);
    }

    // Attention (flash): grid (S/128, B*H)
    {
        dim3 grid(S_LEN / 128, BATCH * NHEAD);
        flash_attn<<<grid, 256, 0, stream>>>(Qb, Kb, Vb, ctxb);
    }

    // Output projection: ctx@Wo+bo -> f32
    {
        dim3 grid(DMODEL / 128, ROWS / 128);
        gemm_bf16<false, true, false><<<grid, 256, 0, stream>>>(
            ctxb, Wob, bo, attn, nullptr, ROWS, DMODEL, DMODEL);
    }

    // src2 = LN(src + attn_out), plus bf16 copy
    add_layernorm<<<ROWS, 256, 0, stream>>>(src, attn, g1, be1, src2, src2b);

    // FFN up: relu(src2@W1+b1) -> bf16 (8192x4096)
    {
        dim3 grid(FFDIM / 128, ROWS / 128);
        gemm_bf16<true, false, true><<<grid, 256, 0, stream>>>(
            src2b, W1b, b1, nullptr, h1b, ROWS, FFDIM, DMODEL);
    }
    // FFN down: h@W2+b2 -> f32
    {
        dim3 grid(DMODEL / 128, ROWS / 128);
        gemm_bf16<false, true, false><<<grid, 256, 0, stream>>>(
            h1b, W2b, b2, ff, nullptr, ROWS, DMODEL, FFDIM);
    }

    // out = LN(src2 + ff)
    add_layernorm<<<ROWS, 256, 0, stream>>>(src2, ff, g2, be2, (float*)d_out, nullptr);
}
